// GAT_63496796504812
// MI455X (gfx1250) — matrix operand order repaired and checked by reading, not compiled, not run on hardware
//
#include <hip/hip_runtime.h>
#include <hip/hip_bf16.h>
#include <float.h>

typedef __attribute__((ext_vector_type(16))) _Float16 v16h;
typedef __attribute__((ext_vector_type(8)))  float    v8f;

#define N_NODES   100000
#define N_EDGES   3200000
#define F_IN      256
#define HEADS     8
#define C_OUT     8
#define F_OUT     64            // HEADS * C_OUT
#define NEG_SLOPE 0.2f

// ---------------------------------------------------------------------------
// Init: out <- broadcast(bias), m <- -FLT_MAX, s <- 0   (buffers are poisoned)
// ---------------------------------------------------------------------------
__global__ void gat_init(float* __restrict__ out, const float* __restrict__ bias,
                         float* __restrict__ m, float* __restrict__ s) {
    int idx = blockIdx.x * blockDim.x + threadIdx.x;
    if (idx < N_NODES * F_OUT) out[idx] = bias[idx & (F_OUT - 1)];
    if (idx < N_NODES * HEADS) { m[idx] = -FLT_MAX; s[idx] = 0.0f; }
}

// ---------------------------------------------------------------------------
// Fused GEMM (h = x @ W via v_wmma_f32_16x16x32_f16) + attention epilogue:
//   a_src[n,h] = sum_c h[n,h,c]*att_src[h,c],  a_dst likewise.
// 256 threads = 8 waves; each wave owns a 16-row M tile; block covers 128 rows.
// W (256x64) staged transposed in LDS as f16.
// ---------------------------------------------------------------------------
__global__ __launch_bounds__(256) void gat_gemm(
    const float* __restrict__ x, const float* __restrict__ W,
    const float* __restrict__ att_src, const float* __restrict__ att_dst,
    float* __restrict__ h, float* __restrict__ asrc, float* __restrict__ adst) {

    __shared__ _Float16 WT[F_OUT][F_IN + 2];   // transposed, padded (~32.3 KB)

    const int tid = threadIdx.x;
    for (int i = tid; i < F_IN * F_OUT; i += 256) {
        int k = i >> 6, n = i & 63;
        WT[n][k] = (_Float16)W[i];
    }
    __syncthreads();

    const int lane = tid & 31;
    const int wave = tid >> 5;
    const int g    = lane >> 4;      // half-wave group (K sub-block select)
    const int hl   = lane & 15;      // M (for A) / N (for B,D) index
    const int m0   = blockIdx.x * 128 + wave * 16;

    const int row  = m0 + hl;
    const int rowc = row < N_NODES ? row : (N_NODES - 1);  // clamp reads
    const float* xr = x + (size_t)rowc * F_IN;

    v8f acc[4] = {};

    for (int kc = 0; kc < F_IN; kc += 32) {
        __builtin_prefetch(xr + kc + 32, 0, 1);   // global_prefetch_b8 next chunk
        // A fragment: lane holds row `row`; halves 0..7 -> K = kc+g*8+(i&7),
        //             halves 8..15 -> K = kc+16+g*8+(i&7)   (ISA 7.12.2)
        v16h a;
#pragma unroll
        for (int i = 0; i < 16; ++i) {
            int k = kc + g * 8 + ((i & 8) ? 16 : 0) + (i & 7);
            a[i] = (_Float16)xr[k];
        }
#pragma unroll
        for (int nt = 0; nt < 4; ++nt) {
            // B fragment: lane holds column nt*16+hl; half i -> K = kc+g*8+(i&8?16:0)+(i&7)
            v16h b;
#pragma unroll
            for (int i = 0; i < 16; ++i)
                b[i] = WT[nt * 16 + hl][kc + g * 8 + ((i & 8) ? 16 : 0) + (i & 7)];
            acc[nt] = __builtin_amdgcn_wmma_f32_16x16x32_f16(
                false, a, false, b, (short)0, acc[nt], false, false);
        }
    }

    // Epilogue: store h, reduce per-head attention scalars.
    // D layout: VGPR r, lanes 0-15 -> M=r, N=lane; lanes 16-31 -> M=8+r.
#pragma unroll
    for (int nt = 0; nt < 4; ++nt) {
#pragma unroll
        for (int r = 0; r < 8; ++r) {
            const int mrow = m0 + r + g * 8;
            const int col  = nt * 16 + hl;
            const float v  = acc[nt][r];
            if (mrow < N_NODES) h[(size_t)mrow * F_OUT + col] = v;

            const int head = col >> 3, c = col & 7;
            float vs = v * att_src[head * C_OUT + c];
            float vd = v * att_dst[head * C_OUT + c];
            // butterfly over the 8 lanes sharing (row, head)
            vs += __shfl_xor(vs, 1); vd += __shfl_xor(vd, 1);
            vs += __shfl_xor(vs, 2); vd += __shfl_xor(vd, 2);
            vs += __shfl_xor(vs, 4); vd += __shfl_xor(vd, 4);
            if ((lane & 7) == 0 && mrow < N_NODES) {
                asrc[mrow * HEADS + head] = vs;
                adst[mrow * HEADS + head] = vd;
            }
        }
    }
}

// ---------------------------------------------------------------------------
// Edge phase helpers
// ---------------------------------------------------------------------------
__device__ __forceinline__ float leaky(float v) {
    return v > 0.0f ? v : NEG_SLOPE * v;
}

// sign-aware float atomic max (int max for >=0, uint min for <0)
__device__ __forceinline__ void atomicMaxF(float* addr, float val) {
    if (val >= 0.0f) atomicMax((int*)addr, __float_as_int(val));
    else             atomicMin((unsigned int*)addr, (unsigned int)__float_as_int(val));
}

// Pass 1: segment max of leaky-relu logits over incoming edges of dst
__global__ void gat_edge_max(const int* __restrict__ ei,
                             const float* __restrict__ asrc,
                             const float* __restrict__ adst,
                             float* __restrict__ m) {
    long long idx = (long long)blockIdx.x * blockDim.x + threadIdx.x;
    if (idx >= (long long)N_EDGES * HEADS) return;
    const int e  = (int)(idx >> 3);
    const int hh = (int)(idx & 7);
    const int src = ei[e], dst = ei[N_EDGES + e];
    const float l = leaky(asrc[src * HEADS + hh] + adst[dst * HEADS + hh]);
    atomicMaxF(&m[dst * HEADS + hh], l);
}

// Pass 2: segment sum of exp(logit - max)
__global__ void gat_edge_sum(const int* __restrict__ ei,
                             const float* __restrict__ asrc,
                             const float* __restrict__ adst,
                             const float* __restrict__ m,
                             float* __restrict__ s) {
    long long idx = (long long)blockIdx.x * blockDim.x + threadIdx.x;
    if (idx >= (long long)N_EDGES * HEADS) return;
    const int e  = (int)(idx >> 3);
    const int hh = (int)(idx & 7);
    const int src = ei[e], dst = ei[N_EDGES + e];
    const float l = leaky(asrc[src * HEADS + hh] + adst[dst * HEADS + hh]);
    atomicAdd(&s[dst * HEADS + hh], expf(l - m[dst * HEADS + hh]));
}

// Pass 3: out[dst] += alpha * h[src]   (8 features per (edge,head) thread)
__global__ void gat_scatter(const int* __restrict__ ei,
                            const float* __restrict__ asrc,
                            const float* __restrict__ adst,
                            const float* __restrict__ m,
                            const float* __restrict__ s,
                            const float* __restrict__ h,
                            float* __restrict__ out) {
    long long idx = (long long)blockIdx.x * blockDim.x + threadIdx.x;
    if (idx >= (long long)N_EDGES * HEADS) return;
    const int e  = (int)(idx >> 3);
    const int hh = (int)(idx & 7);
    const int src = ei[e], dst = ei[N_EDGES + e];
    const float l = leaky(asrc[src * HEADS + hh] + adst[dst * HEADS + hh]);
    const float alpha = expf(l - m[dst * HEADS + hh]) /
                        (s[dst * HEADS + hh] + 1e-16f);
    const float4* hs = (const float4*)(h + (size_t)src * F_OUT + hh * C_OUT);
    const float4 f0 = hs[0], f1 = hs[1];           // global_load_b128 x2
    float* o = out + (size_t)dst * F_OUT + hh * C_OUT;
    atomicAdd(o + 0, alpha * f0.x); atomicAdd(o + 1, alpha * f0.y);
    atomicAdd(o + 2, alpha * f0.z); atomicAdd(o + 3, alpha * f0.w);
    atomicAdd(o + 4, alpha * f1.x); atomicAdd(o + 5, alpha * f1.y);
    atomicAdd(o + 6, alpha * f1.z); atomicAdd(o + 7, alpha * f1.w);
}

// ---------------------------------------------------------------------------
extern "C" void kernel_launch(void* const* d_in, const int* in_sizes, int n_in,
                              void* d_out, int out_size, void* d_ws, size_t ws_size,
                              hipStream_t stream) {
    const float* x       = (const float*)d_in[0];
    const int*   ei      = (const int*)  d_in[1];   // [2, E] (src row, dst row)
    const float* W       = (const float*)d_in[2];
    const float* att_src = (const float*)d_in[3];
    const float* att_dst = (const float*)d_in[4];
    const float* bias    = (const float*)d_in[5];
    float* out = (float*)d_out;

    // workspace carve: h[N,64] | a_src[N,8] | a_dst[N,8] | m[N,8] | s[N,8]
    float* h    = (float*)d_ws;
    float* asrc = h    + (size_t)N_NODES * F_OUT;
    float* adst = asrc + (size_t)N_NODES * HEADS;
    float* mbuf = adst + (size_t)N_NODES * HEADS;
    float* sbuf = mbuf + (size_t)N_NODES * HEADS;

    gat_init<<<(N_NODES * F_OUT + 255) / 256, 256, 0, stream>>>(out, bias, mbuf, sbuf);
    gat_gemm<<<(N_NODES + 127) / 128, 256, 0, stream>>>(x, W, att_src, att_dst,
                                                        h, asrc, adst);
    const long long eh = (long long)N_EDGES * HEADS;
    const int eblocks = (int)((eh + 255) / 256);
    gat_edge_max<<<eblocks, 256, 0, stream>>>(ei, asrc, adst, mbuf);
    gat_edge_sum<<<eblocks, 256, 0, stream>>>(ei, asrc, adst, mbuf, sbuf);
    gat_scatter <<<eblocks, 256, 0, stream>>>(ei, asrc, adst, mbuf, sbuf, h, out);
}
